// MultiHeadAttentionRoPE_79757542687112
// MI455X (gfx1250) — matrix operand order repaired and checked by reading, not compiled, not run on hardware
//
#include <hip/hip_runtime.h>
#include <hip/hip_bf16.h>

// Problem constants (reference: B,S,D,H = 4,2048,1024,16; HD=64)
#define PB 4
#define PS 2048
#define PD 1024
#define PH 16
#define PHD 64

typedef __attribute__((ext_vector_type(16))) _Float16 v16h;
typedef __attribute__((ext_vector_type(8)))  _Float16 v8h;
typedef __attribute__((ext_vector_type(8)))  float    v8f;

// Build a v16h WMMA fragment from two 16-byte chunks (ISA fragment layout:
// per lane, 8 halves at koff and 8 halves at koff+16).
__device__ __forceinline__ v16h load16h(const _Float16* p0, const _Float16* p1) {
  v8h a = *(const v8h*)p0;
  v8h b = *(const v8h*)p1;
  return __builtin_shufflevector(a, b, 0,1,2,3,4,5,6,7,8,9,10,11,12,13,14,15);
}

__device__ __forceinline__ v8f wmma_f16(v16h a, v16h b, v8f c) {
  return __builtin_amdgcn_wmma_f32_16x16x32_f16(false, a, false, b, (short)0, c, false, false);
}

// Butterfly reduction within each 16-lane row via DPP row_xmask (DPP16:
// dpp_ctrl = 0x160 | xor_mask). Stays inside lanes 0-15 / 16-31 halves.
template <int MASK>
__device__ __forceinline__ float dpp_xor_f(float x) {
  int xi = __builtin_bit_cast(int, x);
  int r  = __builtin_amdgcn_update_dpp(xi, xi, 0x160 | MASK, 0xf, 0xf, true);
  return __builtin_bit_cast(float, r);
}
__device__ __forceinline__ float rowmax16(float x) {
  x = fmaxf(x, dpp_xor_f<1>(x));
  x = fmaxf(x, dpp_xor_f<2>(x));
  x = fmaxf(x, dpp_xor_f<4>(x));
  x = fmaxf(x, dpp_xor_f<8>(x));
  return x;
}
__device__ __forceinline__ float rowsum16(float x) {
  x += dpp_xor_f<1>(x);
  x += dpp_xor_f<2>(x);
  x += dpp_xor_f<4>(x);
  x += dpp_xor_f<8>(x);
  return x;
}

// Async global->LDS copy, 16 bytes per lane (CDNA5 GLOBAL_LOAD_ASYNC_TO_LDS,
// tracked by ASYNCcnt). lds_off = byte offset inside the workgroup's LDS
// (low 32 bits of the generic shared-pointer).
__device__ __forceinline__ void async_b128(unsigned lds_off, const void* g) {
  asm volatile("global_load_async_to_lds_b128 %0, %1, off"
               :: "v"(lds_off), "v"(g) : "memory");
}

// ---------------------------------------------------------------------------
// f32 -> f16 convert
// ---------------------------------------------------------------------------
__global__ void cvt_f32_f16(const float* __restrict__ in, _Float16* __restrict__ out, int n) {
  int i = blockIdx.x * blockDim.x + threadIdx.x;
  if (i < n) out[i] = (_Float16)in[i];
}

// ---------------------------------------------------------------------------
// GEMM: out = A(M x K, f16 row-major) @ W(N x K, f16 row-major)^T + bias
// Block = 4 waves = 128x64 tile; wave = 32x64 (two 16-row groups).
// The 64xK weight panel is staged through double-buffered LDS with async
// global->LDS copies (K-stage = 64) and shared by all 4 waves.
// MODE 0: RoPE epilogue (*oscale), f16 store to [B,H,S,64]
// MODE 1: f16 store transposed to [B,H,64,S] (for V)
// MODE 2: f32 + bias store to [M,N] (final output projection)
// ---------------------------------------------------------------------------
template <int MODE>
__global__ __launch_bounds__(128)
void gemm_wmma(const _Float16* __restrict__ A, const _Float16* __restrict__ W,
               const float* __restrict__ bias, void* __restrict__ dst,
               int M, int N, int K, float oscale) {
  __shared__ _Float16 wlds[2][64 * 64];   // 64 n-rows x 64 k-halves per buffer

  const int lt   = threadIdx.x;
  const int wv   = lt >> 5;
  const int lane = lt & 31;
  const int ln   = lane & 15;
  const int hi   = lane >> 4;
  const int koff = hi ? 8 : 0;

  const int ntn = N >> 6;
  const int m0blk = (blockIdx.x / ntn) << 7;   // 128-row block tile
  const int n0    = (blockIdx.x % ntn) << 6;
  const int m0    = m0blk + (wv << 5);         // this wave's 32 rows

  const unsigned wofs[2] = {(unsigned)(uintptr_t)&wlds[0][0],
                            (unsigned)(uintptr_t)&wlds[1][0]};

  // stage 64 n-rows x 64 k-halves (8KB) = 512 16B chunks, 4 per thread
  auto issue = [&](int k0, int sel) {
#pragma unroll
    for (int it = 0; it < 4; ++it) {
      const int c = lt + it * 128;
      const int n = c >> 3, q = c & 7;
      async_b128(wofs[sel] + c * 16,
                 (const char*)(W + (size_t)(n0 + n) * K + k0 + q * 8));
    }
  };

  v8f accs[2][4];
#pragma unroll
  for (int g = 0; g < 2; ++g)
#pragma unroll
    for (int j = 0; j < 4; ++j) accs[g][j] = (v8f){};

  const _Float16* arow0 = A + (size_t)(m0 + ln) * K;
  const _Float16* arow1 = A + (size_t)(m0 + 16 + ln) * K;

  issue(0, 0);
  for (int k0 = 0; k0 < K; k0 += 64) {
    const int sel = (k0 >> 6) & 1;
    if (k0 + 64 < K) {
      issue(k0 + 64, sel ^ 1);
      asm volatile("s_wait_asynccnt 4" ::: "memory");
    } else {
      asm volatile("s_wait_asynccnt 0" ::: "memory");
    }
    __syncthreads();

    const v16h af00 = load16h(arow0 + k0 + koff,      arow0 + k0 + koff + 16);
    const v16h af01 = load16h(arow0 + k0 + 32 + koff, arow0 + k0 + 32 + koff + 16);
    const v16h af10 = load16h(arow1 + k0 + koff,      arow1 + k0 + koff + 16);
    const v16h af11 = load16h(arow1 + k0 + 32 + koff, arow1 + k0 + 32 + koff + 16);

#pragma unroll
    for (int j = 0; j < 4; ++j) {
      const _Float16* wr = &wlds[sel][(size_t)(16 * j + ln) * 64];
      const v16h b0 = load16h(wr + koff,      wr + koff + 16);
      const v16h b1 = load16h(wr + 32 + koff, wr + 32 + koff + 16);
      accs[0][j] = wmma_f16(af00, b0, accs[0][j]);
      accs[0][j] = wmma_f16(af01, b1, accs[0][j]);
      accs[1][j] = wmma_f16(af10, b0, accs[1][j]);
      accs[1][j] = wmma_f16(af11, b1, accs[1][j]);
    }
    __syncthreads();
  }

  // C/D layout: VGPR r, lane l -> row M = r + 8*(l>=16), col N = l%16 per tile
  const int h = n0 >> 6;
#pragma unroll
  for (int g = 0; g < 2; ++g) {
    const int mg = m0 + 16 * g;
    const int b  = mg >> 11;
    const int sb = mg & (PS - 1);

    if (MODE == 0) {
      _Float16* dq = (_Float16*)dst + (((size_t)b * PH + h) * PS) * PHD;
#pragma unroll
      for (int j = 0; j < 2; ++j) {
        const int d = 16 * j + ln;
        const float bias1 = bias[n0 + d];
        const float bias2 = bias[n0 + d + 32];
        const float invf  = __powf(10000.0f, -(float)d * (1.0f / 32.0f));
#pragma unroll
        for (int r = 0; r < 8; ++r) {
          const int s = sb + r + 8 * hi;
          const float ang = (float)s * invf;
          const float c  = __cosf(ang);
          const float sn = __sinf(ang);
          const float x1 = (accs[g][j][r]     + bias1) * oscale;
          const float x2 = (accs[g][j + 2][r] + bias2) * oscale;
          dq[(size_t)s * PHD + d]      = (_Float16)(x1 * c - x2 * sn);
          dq[(size_t)s * PHD + d + 32] = (_Float16)(x1 * sn + x2 * c);
        }
      }
    } else if (MODE == 1) {
      _Float16* dv = (_Float16*)dst + (((size_t)b * PH + h) * PHD) * PS;
#pragma unroll
      for (int j = 0; j < 4; ++j) {
        const int d = 16 * j + ln;
        const float bv = bias[n0 + d];
        v8h pack;
#pragma unroll
        for (int r = 0; r < 8; ++r) pack[r] = (_Float16)(accs[g][j][r] + bv);
        *(v8h*)(dv + (size_t)d * PS + sb + 8 * hi) = pack;
      }
    } else {
      float* o = (float*)dst;
#pragma unroll
      for (int j = 0; j < 4; ++j) {
        const int n = n0 + 16 * j + ln;
        const float bv = bias[n];
#pragma unroll
        for (int r = 0; r < 8; ++r)
          o[(size_t)(mg + r + 8 * hi) * N + n] = accs[g][j][r] + bv;
      }
    }
  }
}

// ---------------------------------------------------------------------------
// Flash attention (causal). One block = 4 waves = 64 query rows of one (b,h).
// K/V tiles are double-buffered in LDS via async global->LDS copies; the 4
// waves share them. q: pre-scaled by 1/sqrt(HD) in the projection epilogue.
// q,k: f16 [B,H,S,64] (RoPE applied). vt: f16 [B,H,64,S]. out: f16 [B,S,D].
// ---------------------------------------------------------------------------
__global__ __launch_bounds__(128)
void flash_attn(const _Float16* __restrict__ q, const _Float16* __restrict__ k,
                const _Float16* __restrict__ vt, _Float16* __restrict__ o_attn) {
  __shared__ _Float16 kbuf[2][32 * 64];   // K tile: 32 kv rows x 64 d
  __shared__ _Float16 vbuf[2][64 * 32];   // Vt tile: 64 d rows x 32 kv
  __shared__ _Float16 plds[4][16 * 32];   // per-wave P staging

  const int lt   = threadIdx.x;
  const int wv   = lt >> 5;
  const int lane = lt & 31;
  const int ln   = lane & 15;
  const int hi   = lane >> 4;
  const int koff = hi ? 8 : 0;

  const int blk = blockIdx.x;
  const int qt  = blk & (PS / 64 - 1);    // 32 query tiles per (b,h)
  const int bh  = blk >> 5;
  const int b   = bh >> 4;
  const int h   = bh & 15;

  const _Float16* kb_ = k  + ((size_t)bh * PS) * PHD;
  const _Float16* vb_ = vt + ((size_t)bh * PHD) * PS;

  const int qt0 = qt << 6;                // block's first query row
  const int qs0 = qt0 + (wv << 4);        // this wave's first query row

  // Q fragments, resident for the whole KV loop (already scaled by 1/8)
  const _Float16* qrow = q + ((size_t)bh * PS + qs0 + ln) * PHD;
  const v16h aq0 = load16h(qrow + koff,      qrow + koff + 16);
  const v16h aq1 = load16h(qrow + 32 + koff, qrow + 32 + koff + 16);

  const unsigned kofs[2] = {(unsigned)(uintptr_t)&kbuf[0][0],
                            (unsigned)(uintptr_t)&kbuf[1][0]};
  const unsigned vofs[2] = {(unsigned)(uintptr_t)&vbuf[0][0],
                            (unsigned)(uintptr_t)&vbuf[1][0]};

  // Cooperative async stage of one 32-wide KV step (4 asyncs per thread).
  auto issue = [&](int kv0, int sel) {
#pragma unroll
    for (int it = 0; it < 2; ++it) {
      const int c = lt + it * 128;        // 256 16-byte chunks per tile
      const int krow = c >> 3, kq = c & 7;
      async_b128(kofs[sel] + c * 16,
                 (const char*)kb_ + ((size_t)(kv0 + krow) * PHD + kq * 8) * 2);
      const int vrow = c >> 2, vq = c & 3;
      async_b128(vofs[sel] + c * 16,
                 (const char*)vb_ + ((size_t)vrow * PS + kv0 + vq * 8) * 2);
    }
  };

  float m_run[8], l_run[8];
#pragma unroll
  for (int r = 0; r < 8; ++r) { m_run[r] = -1e30f; l_run[r] = 0.0f; }
  v8f o0 = {}, o1 = {}, o2 = {}, o3 = {};

  const int nsteps = (qt0 + 64) >> 5;     // block-uniform causal bound
  issue(0, 0);

  for (int i = 0; i < nsteps; ++i) {
    const int kv0 = i << 5;
    const int sel = i & 1;
    if (i + 1 < nsteps) {
      issue(kv0 + 32, sel ^ 1);
      asm volatile("s_wait_asynccnt 4" ::: "memory");   // step i's copies done
    } else {
      asm volatile("s_wait_asynccnt 0" ::: "memory");
    }
    __syncthreads();                      // publish tile across waves

    // scores: S = Q(16x64) @ K(32x64)^T  (fragments from LDS)
    v8f sc0 = {}, sc1 = {};
    {
      const _Float16* kr0 = &kbuf[sel][(size_t)ln * 64];
      const _Float16* kr1 = &kbuf[sel][(size_t)(16 + ln) * 64];
      v16h b00 = load16h(kr0 + koff,      kr0 + koff + 16);
      v16h b01 = load16h(kr0 + 32 + koff, kr0 + 32 + koff + 16);
      v16h b10 = load16h(kr1 + koff,      kr1 + koff + 16);
      v16h b11 = load16h(kr1 + 32 + koff, kr1 + 32 + koff + 16);
      sc0 = wmma_f16(aq0, b00, sc0);
      sc0 = wmma_f16(aq1, b01, sc0);
      sc1 = wmma_f16(aq0, b10, sc1);
      sc1 = wmma_f16(aq1, b11, sc1);
    }

    // online softmax; DPP row_xmask reductions within 16-lane halves
#pragma unroll
    for (int r = 0; r < 8; ++r) {
      const int qr = qs0 + r + 8 * hi;
      float v0 = sc0[r]; if (kv0 + ln > qr)      v0 = -1e30f;
      float v1 = sc1[r]; if (kv0 + 16 + ln > qr) v1 = -1e30f;
      const float mx    = rowmax16(fmaxf(v0, v1));
      const float m_new = fmaxf(m_run[r], mx);
      const float corr  = __expf(m_run[r] - m_new);
      const float p0 = __expf(v0 - m_new);
      const float p1 = __expf(v1 - m_new);
      const float ps = rowsum16(p0 + p1);
      l_run[r] = l_run[r] * corr + ps;
      m_run[r] = m_new;
      o0[r] *= corr; o1[r] *= corr; o2[r] *= corr; o3[r] *= corr;
      // stage P in [M][K] layout (wave-local; DS ops in-order per wave)
      plds[wv][(r + 8 * hi) * 32 + ln]      = (_Float16)p0;
      plds[wv][(r + 8 * hi) * 32 + 16 + ln] = (_Float16)p1;
    }

    // P as A-fragment (16x32)
    const v16h ap = load16h(&plds[wv][ln * 32 + koff],
                            &plds[wv][ln * 32 + koff + 16]);

    // O += P(16x32) @ V(32x64)
    {
      const _Float16* vr0 = &vbuf[sel][(size_t)ln * 32];
      const _Float16* vr1 = &vbuf[sel][(size_t)(16 + ln) * 32];
      const _Float16* vr2 = &vbuf[sel][(size_t)(32 + ln) * 32];
      const _Float16* vr3 = &vbuf[sel][(size_t)(48 + ln) * 32];
      o0 = wmma_f16(ap, load16h(vr0 + koff, vr0 + koff + 16), o0);
      o1 = wmma_f16(ap, load16h(vr1 + koff, vr1 + koff + 16), o1);
      o2 = wmma_f16(ap, load16h(vr2 + koff, vr2 + koff + 16), o2);
      o3 = wmma_f16(ap, load16h(vr3 + koff, vr3 + koff + 16), o3);
    }
    __syncthreads();                      // tile consumed; safe to overwrite
  }

  // normalize + write f16 [b][s][h*64 + d] for the output projection
  v8f oo[4] = {o0, o1, o2, o3};
#pragma unroll
  for (int r = 0; r < 8; ++r) {
    const float inv = 1.0f / l_run[r];
    const int s = qs0 + r + 8 * hi;
    _Float16* orow = o_attn + ((size_t)b * PS + s) * PD + h * PHD;
#pragma unroll
    for (int j = 0; j < 4; ++j)
      orow[16 * j + ln] = (_Float16)(oo[j][r] * inv);
  }
}

// ---------------------------------------------------------------------------
// launch
// ---------------------------------------------------------------------------
extern "C" void kernel_launch(void* const* d_in, const int* in_sizes, int n_in,
                              void* d_out, int out_size, void* d_ws, size_t ws_size,
                              hipStream_t stream) {
  const float* Q  = (const float*)d_in[0];
  const float* K  = (const float*)d_in[1];
  const float* V  = (const float*)d_in[2];
  const float* Wq = (const float*)d_in[3];
  const float* bq = (const float*)d_in[4];
  const float* Wk = (const float*)d_in[5];
  const float* bk = (const float*)d_in[6];
  const float* Wv = (const float*)d_in[7];
  const float* bv = (const float*)d_in[8];
  const float* Wo = (const float*)d_in[9];
  const float* bo = (const float*)d_in[10];
  float* out = (float*)d_out;

  const size_t WB = (size_t)PD * PD;
  const size_t XB = (size_t)PB * PS * PD;
  const size_t HB = (size_t)PB * PH * PS * PHD;

  char* p = (char*)d_ws;
  _Float16* Wqb = (_Float16*)p; p += WB * 2;
  _Float16* Wkb = (_Float16*)p; p += WB * 2;
  _Float16* Wvb = (_Float16*)p; p += WB * 2;
  _Float16* Wob = (_Float16*)p; p += WB * 2;
  _Float16* Xb  = (_Float16*)p; p += XB * 2;  // reused: Q-in, K-in, V-in, attn-out
  _Float16* q2  = (_Float16*)p; p += HB * 2;  // [B,H,S,64] roped, pre-scaled
  _Float16* k2  = (_Float16*)p; p += HB * 2;  // [B,H,S,64] roped
  _Float16* vtb = (_Float16*)p; p += HB * 2;  // [B,H,64,S]

  const int M = PB * PS, N = PD, Kd = PD;
  const int gemm_blocks  = (M / 128) * (N / 64);   // 128x64 block tiles
  const int flash_blocks = PB * PH * (PS / 64);    // 1 block per 64-row q tile

  cvt_f32_f16<<<(int)((WB + 255) / 256), 256, 0, stream>>>(Wq, Wqb, (int)WB);
  cvt_f32_f16<<<(int)((WB + 255) / 256), 256, 0, stream>>>(Wk, Wkb, (int)WB);
  cvt_f32_f16<<<(int)((WB + 255) / 256), 256, 0, stream>>>(Wv, Wvb, (int)WB);
  cvt_f32_f16<<<(int)((WB + 255) / 256), 256, 0, stream>>>(Wo, Wob, (int)WB);

  cvt_f32_f16<<<(int)((XB + 255) / 256), 256, 0, stream>>>(Q, Xb, (int)XB);
  gemm_wmma<0><<<gemm_blocks, 128, 0, stream>>>(Xb, Wqb, bq, q2, M, N, Kd, 0.125f);

  cvt_f32_f16<<<(int)((XB + 255) / 256), 256, 0, stream>>>(K, Xb, (int)XB);
  gemm_wmma<0><<<gemm_blocks, 128, 0, stream>>>(Xb, Wkb, bk, k2, M, N, Kd, 1.0f);

  cvt_f32_f16<<<(int)((XB + 255) / 256), 256, 0, stream>>>(V, Xb, (int)XB);
  gemm_wmma<1><<<gemm_blocks, 128, 0, stream>>>(Xb, Wvb, bv, vtb, M, N, Kd, 1.0f);

  flash_attn<<<flash_blocks, 128, 0, stream>>>(q2, k2, vtb, Xb);

  gemm_wmma<2><<<gemm_blocks, 128, 0, stream>>>(Xb, Wob, bo, out, M, N, Kd, 1.0f);
}
